// FlowODEv3_75797582839849
// MI455X (gfx1250) — matrix operand order repaired and checked by reading, not compiled, not run on hardware
//
#include <hip/hip_runtime.h>
#include <hip/hip_bf16.h>
#include <math.h>
#include <stdint.h>

// ---------------------------------------------------------------------------
// FlowODE forward on gfx1250 (CDNA5).
//   * All GEMMs run on v_wmma_f32_16x16x32_bf16 (wave32).
//   * The two big plain GEMMs (z0 = x P^T and logits = h W_lm^T) use the
//     Tensor Data Mover: double-buffered LDS tiles staged with
//     tensor_load_to_lds (TENSORcnt) and consumed via ds_load fragments.
//   * The ODE-step MLP GEMMs keep the fused bias/GELU/residual epilogue path.
// ---------------------------------------------------------------------------

typedef __bf16 bf16_t;
typedef __attribute__((ext_vector_type(16))) __bf16 v16bf;
typedef __attribute__((ext_vector_type(8)))  __bf16 v8bf;
typedef __attribute__((ext_vector_type(8)))  float  v8f;
typedef __attribute__((ext_vector_type(4)))  unsigned int v4u;
typedef __attribute__((ext_vector_type(8)))  unsigned int v8u;

#define D_DIM  2048
#define R_DIM  256
#define T_DIM  64
#define ZT_DIM 320   // R + TDIM
#define HID    768   // R*3
#define NSTEPS 16
#define VOCAB  32000

// ---------------------------------------------------------------------------
// Fragment loaders (row-major [rows, K] bf16 source, works for global or LDS).
// A fragment (16x32): lane l: row=l&15, half=l>>4;
//   K = half*16..+15 contiguous (two 16B chunks), matching B.
// B fragment (32x16 dense, same k order as A, not the ISA table order): lane l: col=l&15, half=l>>4;
//   K = half*16..+15 contiguous (one 32B chunk).
// ---------------------------------------------------------------------------
template <typename PtrT>
__device__ inline v16bf load_fragA(PtrT base, int ld, int row, int kb, int half, int l15)
{
    PtrT p = base + (size_t)(row + l15) * ld + kb + half * 16;
    v8bf lo = *(const v8bf*)p;
    v8bf hi = *(const v8bf*)(p + 8);
    return __builtin_shufflevector(lo, hi, 0,1,2,3,4,5,6,7,8,9,10,11,12,13,14,15);
}

template <typename PtrT>
__device__ inline v16bf load_fragB(PtrT base, int ld, int col, int kb, int half, int l15)
{
    PtrT p = base + (size_t)(col + l15) * ld + kb + half * 16;
    return *(const v16bf*)p;
}

// ---------------------------------------------------------------------------
// TDM: DMA a 2-D tile (rows x 32 bf16) from a row-major [*, ld] matrix into
// LDS at byte offset lds_addr. D# built per ISA ch.8:
//   group0: count=1 | lds_addr | global_addr[56:0] | type=2
//   group1: data_size=1(2B), tensor_dim0=32, tensor_dim1=rows,
//           tile_dim0=32, tile_dim1=rows, tensor_dim0_stride=ld
// 2-D tensor -> descriptor groups 2/3 omitted (VADDR2/3 = NULL).
// ---------------------------------------------------------------------------
__device__ inline void tdm_load_tile(unsigned lds_addr, const bf16_t* gptr,
                                     unsigned rows, unsigned ld)
{
    const unsigned long long ga = (unsigned long long)(uintptr_t)gptr;
    v4u g0;
    g0.x = 1u;                                            // count=1 (valid user D#)
    g0.y = lds_addr;                                      // LDS byte address
    g0.z = (unsigned)(ga & 0xFFFFFFFFu);                  // global_addr[31:0]
    g0.w = (unsigned)((ga >> 32) & 0x01FFFFFFu) | (2u << 30);  // addr[56:32] | type=2
    v8u g1;
    g1.s0 = 0x00010000u;          // workgroup_mask=0, data_size=1 (2 bytes)
    g1.s1 = (32u << 16);          // tensor_dim0[15:0]=32 (in d1[31:16])
    g1.s2 = (rows << 16);         // tensor_dim0[31:16]=0 | tensor_dim1[15:0]=rows
    g1.s3 = (32u << 16);          // tensor_dim1[31:16]=0 | tile_dim0=32
    g1.s4 = rows;                 // tile_dim1=rows, tile_dim2=0
    g1.s5 = ld;                   // tensor_dim0_stride[31:0]
    g1.s6 = 0u;                   // stride0 hi | stride1 lo (unused for 2-D)
    g1.s7 = 0u;
    asm volatile("tensor_load_to_lds %0, %1" :: "s"(g0), "s"(g1) : "memory");
}

// ---------------------------------------------------------------------------
// TDM-staged WMMA GEMM: C[M,N] = A[M,K] * Bt[N,K]^T, f32 out, no epilogue.
// Block = 4 waves = 64M x 128N tile; wave (2x2) computes 32M x 64N.
// Double-buffered LDS K-slabs (A: 64x32, B: 128x32) loaded by wave 0 via TDM.
// Requires M%64==0, N%128==0, K%32==0 (true for all uses here) -> no tails,
// EXEC stays all-ones for WMMA.
// ---------------------------------------------------------------------------
__launch_bounds__(128)
__global__ void gemm_bf16_wmma_tdm(const bf16_t* __restrict__ A,
                                   const bf16_t* __restrict__ Bt,
                                   float* __restrict__ C,
                                   int M, int N, int K,
                                   int lda, int ldb, int ldc)
{
    __shared__ bf16_t As[2][64 * 32];    // 2 x 4 KB
    __shared__ bf16_t Bs[2][128 * 32];   // 2 x 8 KB

    const int wave = threadIdx.x >> 5;
    const int lane = threadIdx.x & 31;
    const int half = lane >> 4;
    const int l15  = lane & 15;

    const int mt = M >> 6;                       // 64-row block tiles
    const int bm = blockIdx.x % mt;
    const int bn = blockIdx.x / mt;
    const int wr = (wave >> 1) * 32;             // wave row offset in block tile
    const int wc = (wave & 1) * 64;              // wave col offset in block tile

    const bf16_t* Ag = A  + (size_t)(bm * 64)  * lda;
    const bf16_t* Bg = Bt + (size_t)(bn * 128) * ldb;

    const unsigned ldsA[2] = { (unsigned)(uintptr_t)&As[0][0], (unsigned)(uintptr_t)&As[1][0] };
    const unsigned ldsB[2] = { (unsigned)(uintptr_t)&Bs[0][0], (unsigned)(uintptr_t)&Bs[1][0] };

    if (wave == 0) {                              // wave-uniform; TDM ignores EXEC
        tdm_load_tile(ldsA[0], Ag, 64u, (unsigned)lda);
        tdm_load_tile(ldsB[0], Bg, 128u, (unsigned)ldb);
        __builtin_amdgcn_s_wait_tensorcnt(0);
    }
    __syncthreads();

    v8f acc[2][4] = {};
    int cur = 0;
    for (int kb = 0; kb < K; kb += 32) {
        const int nxt = cur ^ 1;
        const bool more = (kb + 32) < K;
        if (wave == 0 && more) {                  // prefetch next K-slab via TDM
            tdm_load_tile(ldsA[nxt], Ag + (kb + 32), 64u, (unsigned)lda);
            tdm_load_tile(ldsB[nxt], Bg + (kb + 32), 128u, (unsigned)ldb);
        }

        v16bf a[2], b[4];
#pragma unroll
        for (int i = 0; i < 2; ++i)
            a[i] = load_fragA<const bf16_t*>(&As[cur][0], 32, wr + i * 16, 0, half, l15);
#pragma unroll
        for (int j = 0; j < 4; ++j)
            b[j] = load_fragB<const bf16_t*>(&Bs[cur][0], 32, wc + j * 16, 0, half, l15);
#pragma unroll
        for (int i = 0; i < 2; ++i)
#pragma unroll
            for (int j = 0; j < 4; ++j)
                acc[i][j] = __builtin_amdgcn_wmma_f32_16x16x32_bf16(
                    false, a[i], false, b[j], (short)0, acc[i][j], false, false);

        if (wave == 0 && more) __builtin_amdgcn_s_wait_tensorcnt(0);
        __syncthreads();
        cur = nxt;
    }

#pragma unroll
    for (int i = 0; i < 2; ++i)
#pragma unroll
        for (int j = 0; j < 4; ++j) {
            const int n = bn * 128 + wc + j * 16 + l15;
#pragma unroll
            for (int r = 0; r < 8; ++r) {
                const int m = bm * 64 + wr + i * 16 + half * 8 + r;
                C[(size_t)m * ldc + n] = acc[i][j][r];
            }
        }
}

// ---------------------------------------------------------------------------
// Generic direct-load WMMA GEMM with fused epilogue (bias -> GELU -> +Cin),
// f32 and/or bf16 output. Used for the ODE-step MLPs and x_out residual GEMM.
// ---------------------------------------------------------------------------
__launch_bounds__(128)
__global__ void gemm_bf16_wmma(const bf16_t* __restrict__ A, int lda,
                               const bf16_t* __restrict__ Bt, int ldb,
                               float* __restrict__ Cf, bf16_t* __restrict__ Cb, int ldc,
                               const float* __restrict__ Cin,
                               const float* __restrict__ bias, int act,
                               int M, int N, int K)
{
    const int lane = threadIdx.x & 31;
    const int wid  = blockIdx.x * (blockDim.x >> 5) + (threadIdx.x >> 5);
    const int mt   = M >> 5;
    const int nt   = N >> 6;
    if (wid >= mt * nt) return;                   // wave-uniform exit
    const int m0   = (wid % mt) << 5;
    const int n0   = (wid / mt) << 6;
    const int half = lane >> 4;
    const int l15  = lane & 15;

    v8f acc[2][4] = {};
    for (int kb = 0; kb < K; kb += 32) {
        v16bf a[2], b[4];
#pragma unroll
        for (int i = 0; i < 2; ++i)
            a[i] = load_fragA<const bf16_t*>(A, lda, m0 + i * 16, kb, half, l15);
#pragma unroll
        for (int j = 0; j < 4; ++j)
            b[j] = load_fragB<const bf16_t*>(Bt, ldb, n0 + j * 16, kb, half, l15);
#pragma unroll
        for (int i = 0; i < 2; ++i)
#pragma unroll
            for (int j = 0; j < 4; ++j)
                acc[i][j] = __builtin_amdgcn_wmma_f32_16x16x32_bf16(
                    false, a[i], false, b[j], (short)0, acc[i][j], false, false);
    }

#pragma unroll
    for (int i = 0; i < 2; ++i)
#pragma unroll
        for (int j = 0; j < 4; ++j) {
            const int n = n0 + j * 16 + l15;
            const float bv = bias ? bias[n] : 0.0f;
#pragma unroll
            for (int r = 0; r < 8; ++r) {
                const int m = m0 + i * 16 + half * 8 + r;
                float v = acc[i][j][r] + bv;
                if (act) v = 0.5f * v * (1.0f + erff(v * 0.7071067811865475f)); // exact GELU
                const size_t idx = (size_t)m * ldc + n;
                if (Cin) v += Cin[idx];
                if (Cf)  Cf[idx] = v;
                if (Cb)  Cb[idx] = (bf16_t)v;
            }
        }
}

// -------------------------- support kernels --------------------------------

__global__ void cvt_f32_bf16_kernel(const float* __restrict__ in, bf16_t* __restrict__ out, size_t n)
{
    size_t i = (size_t)blockIdx.x * blockDim.x + threadIdx.x;
    const size_t stride = (size_t)gridDim.x * blockDim.x;
    for (; i < n; i += stride) out[i] = (bf16_t)in[i];
}

__global__ void gather_kernel(const int* __restrict__ tokens, const float* __restrict__ embed,
                              float* __restrict__ x, bf16_t* __restrict__ xb)
{
    const int row = blockIdx.x;
    const float* src = embed + (size_t)tokens[row] * D_DIM;
    float*  xr  = x  + (size_t)row * D_DIM;
    bf16_t* xbr = xb + (size_t)row * D_DIM;
    for (int c = threadIdx.x; c < D_DIM; c += blockDim.x) {
        const float v = src[c];
        xr[c] = v; xbr[c] = (bf16_t)v;
    }
}

__global__ void pack_zt_kernel(const float* __restrict__ z, const bf16_t* __restrict__ te,
                               bf16_t* __restrict__ Aout)
{
    const int row = blockIdx.x;
    const float* zr = z + (size_t)row * R_DIM;
    bf16_t* ar = Aout + (size_t)row * ZT_DIM;
    for (int c = threadIdx.x; c < ZT_DIM; c += blockDim.x)
        ar[c] = (c < R_DIM) ? (bf16_t)zr[c] : te[c - R_DIM];
}

__global__ void pack_z1_kernel(const float* __restrict__ z, const float* __restrict__ d1,
                               const bf16_t* __restrict__ te, bf16_t* __restrict__ Aout)
{
    const int row = blockIdx.x;
    const float* zr = z  + (size_t)row * R_DIM;
    const float* dr = d1 + (size_t)row * R_DIM;
    bf16_t* ar = Aout + (size_t)row * ZT_DIM;
    for (int c = threadIdx.x; c < ZT_DIM; c += blockDim.x)
        ar[c] = (c < R_DIM) ? (bf16_t)(zr[c] + dr[c]) : te[c - R_DIM];
}

__global__ void z_update_kernel(float* __restrict__ z, const float* __restrict__ d1,
                                const float* __restrict__ d2, const float* __restrict__ log_tau,
                                int s, int n)
{
    const float tau = expf(log_tau[s]);
    for (int i = blockIdx.x * blockDim.x + threadIdx.x; i < n; i += gridDim.x * blockDim.x)
        z[i] += tau * (d1[i] + d2[i]) * 0.01f;
}

__global__ void rmsnorm_kernel(const float* __restrict__ xo, const float* __restrict__ w,
                               bf16_t* __restrict__ hb)
{
    __shared__ float red[256];
    const int row = blockIdx.x;
    const float* xr = xo + (size_t)row * D_DIM;
    float s = 0.0f;
    for (int c = threadIdx.x; c < D_DIM; c += blockDim.x) { const float v = xr[c]; s += v * v; }
    red[threadIdx.x] = s;
    __syncthreads();
    for (int st = 128; st > 0; st >>= 1) {
        if ((int)threadIdx.x < st) red[threadIdx.x] += red[threadIdx.x + st];
        __syncthreads();
    }
    const float scale = rsqrtf(red[0] / (float)D_DIM + 1.1920929e-07f); // FLT_EPSILON
    bf16_t* hr = hb + (size_t)row * D_DIM;
    for (int c = threadIdx.x; c < D_DIM; c += blockDim.x)
        hr[c] = (bf16_t)(xr[c] * scale * w[c]);
}

// ------------------------------ host side ----------------------------------

static inline void run_gemm(hipStream_t stream,
                            const bf16_t* A, int lda, const bf16_t* Bt, int ldb,
                            float* Cf, bf16_t* Cb, int ldc,
                            const float* Cin, const float* bias, int act,
                            int M, int N, int K)
{
    const int tiles = (M >> 5) * (N >> 6);
    const int blocks = (tiles + 3) / 4;
    gemm_bf16_wmma<<<blocks, 128, 0, stream>>>(A, lda, Bt, ldb, Cf, Cb, ldc, Cin, bias, act, M, N, K);
}

static inline void run_gemm_tdm(hipStream_t stream,
                                const bf16_t* A, const bf16_t* Bt, float* C,
                                int M, int N, int K, int lda, int ldb, int ldc)
{
    const int blocks = (M >> 6) * (N >> 7);
    gemm_bf16_wmma_tdm<<<blocks, 128, 0, stream>>>(A, Bt, C, M, N, K, lda, ldb, ldc);
}

extern "C" void kernel_launch(void* const* d_in, const int* in_sizes, int n_in,
                              void* d_out, int out_size, void* d_ws, size_t ws_size,
                              hipStream_t stream)
{
    (void)n_in; (void)out_size; (void)ws_size;

    const int*   tokens    = (const int*)  d_in[0];
    const float* embed_w   = (const float*)d_in[1];
    const float* lm_head_w = (const float*)d_in[2];
    const float* norm_w    = (const float*)d_in[3];
    const float* P         = (const float*)d_in[4];
    const float* Q         = (const float*)d_in[5];
    const float* w11       = (const float*)d_in[6];
    const float* b11       = (const float*)d_in[7];
    const float* w12       = (const float*)d_in[8];
    const float* b12       = (const float*)d_in[9];
    const float* w21       = (const float*)d_in[10];
    const float* b21       = (const float*)d_in[11];
    const float* w22       = (const float*)d_in[12];
    const float* b22       = (const float*)d_in[13];
    const float* time_embs = (const float*)d_in[14];
    const float* log_tau   = (const float*)d_in[15];
    float* out = (float*)d_out;

    const int NT = in_sizes[0];  // B*T = 4096

    char* ws = (char*)d_ws; size_t off = 0;
    auto carve = [&](size_t bytes) -> char* {
        char* p = ws + off; off += (bytes + 255) & ~(size_t)255; return p;
    };
    float*  x    = (float*) carve((size_t)NT * D_DIM * 4);
    bf16_t* xb   = (bf16_t*)carve((size_t)NT * D_DIM * 2);
    float*  z    = (float*) carve((size_t)NT * R_DIM * 4);
    bf16_t* zb   = (bf16_t*)carve((size_t)NT * R_DIM * 2);
    float*  d1   = (float*) carve((size_t)NT * R_DIM * 4);
    float*  d2   = (float*) carve((size_t)NT * R_DIM * 4);
    bf16_t* A1   = (bf16_t*)carve((size_t)NT * ZT_DIM * 2);
    bf16_t* G1   = (bf16_t*)carve((size_t)NT * HID * 2);
    bf16_t* hb   = (bf16_t*)carve((size_t)NT * D_DIM * 2);
    bf16_t* Pb   = (bf16_t*)carve((size_t)R_DIM * D_DIM * 2);
    bf16_t* Qb   = (bf16_t*)carve((size_t)D_DIM * R_DIM * 2);
    bf16_t* w11b = (bf16_t*)carve((size_t)HID * ZT_DIM * 2);
    bf16_t* w21b = (bf16_t*)carve((size_t)HID * ZT_DIM * 2);
    bf16_t* w12b = (bf16_t*)carve((size_t)R_DIM * HID * 2);
    bf16_t* w22b = (bf16_t*)carve((size_t)R_DIM * HID * 2);
    bf16_t* teb  = (bf16_t*)carve((size_t)NSTEPS * T_DIM * 2);
    bf16_t* Lb   = (bf16_t*)carve((size_t)VOCAB * D_DIM * 2);

    // ---- one-shot f32 -> bf16 conversions ----
    cvt_f32_bf16_kernel<<<256, 256, 0, stream>>>(P,   Pb,   (size_t)R_DIM * D_DIM);
    cvt_f32_bf16_kernel<<<256, 256, 0, stream>>>(Q,   Qb,   (size_t)D_DIM * R_DIM);
    cvt_f32_bf16_kernel<<<256, 256, 0, stream>>>(w11, w11b, (size_t)HID * ZT_DIM);
    cvt_f32_bf16_kernel<<<256, 256, 0, stream>>>(w21, w21b, (size_t)HID * ZT_DIM);
    cvt_f32_bf16_kernel<<<256, 256, 0, stream>>>(w12, w12b, (size_t)R_DIM * HID);
    cvt_f32_bf16_kernel<<<256, 256, 0, stream>>>(w22, w22b, (size_t)R_DIM * HID);
    cvt_f32_bf16_kernel<<<1,   256, 0, stream>>>(time_embs, teb, (size_t)NSTEPS * T_DIM);
    cvt_f32_bf16_kernel<<<8192, 256, 0, stream>>>(lm_head_w, Lb, (size_t)VOCAB * D_DIM);

    // ---- x = embed[tokens] ----
    gather_kernel<<<NT, 256, 0, stream>>>(tokens, embed_w, x, xb);

    // ---- z0 = x @ P^T (TDM-staged WMMA) ----
    run_gemm_tdm(stream, xb, Pb, z, NT, R_DIM, D_DIM, D_DIM, D_DIM, R_DIM);

    // ---- 16 ODE steps ----
    for (int s = 0; s < NSTEPS; ++s) {
        const bf16_t* te = teb + (size_t)s * T_DIM;
        pack_zt_kernel<<<NT, 256, 0, stream>>>(z, te, A1);
        run_gemm(stream, A1, ZT_DIM, w11b, ZT_DIM, nullptr, G1, HID, nullptr, b11, 1, NT, HID, ZT_DIM);
        run_gemm(stream, G1, HID, w12b, HID, d1, nullptr, R_DIM, nullptr, b12, 0, NT, R_DIM, HID);
        pack_z1_kernel<<<NT, 256, 0, stream>>>(z, d1, te, A1);
        run_gemm(stream, A1, ZT_DIM, w21b, ZT_DIM, nullptr, G1, HID, nullptr, b21, 1, NT, HID, ZT_DIM);
        run_gemm(stream, G1, HID, w22b, HID, d2, nullptr, R_DIM, nullptr, b22, 0, NT, R_DIM, HID);
        z_update_kernel<<<256, 256, 0, stream>>>(z, d1, d2, log_tau, s, NT * R_DIM);
    }

    // ---- x_out = x + z @ Q^T (in-place residual add) ----
    cvt_f32_bf16_kernel<<<256, 256, 0, stream>>>(z, zb, (size_t)NT * R_DIM);
    run_gemm(stream, zb, R_DIM, Qb, R_DIM, x, nullptr, D_DIM, x, nullptr, 0, NT, D_DIM, R_DIM);

    // ---- RMSNorm -> bf16 h ----
    rmsnorm_kernel<<<NT, 256, 0, stream>>>(x, norm_w, hb);

    // ---- logits = h @ lm_head^T (TDM-staged WMMA) ----
    run_gemm_tdm(stream, hb, Lb, out, NT, VOCAB, D_DIM, D_DIM, D_DIM, VOCAB);
}
